// SynthesisLayer2_32435593020194
// MI455X (gfx1250) — matrix operand order repaired and checked by reading, not compiled, not run on hardware
//
#include <hip/hip_runtime.h>
#include <hip/hip_bf16.h>

typedef __attribute__((ext_vector_type(16))) _Float16 v16h;
typedef __attribute__((ext_vector_type(8)))  _Float16 v8h;
typedef __attribute__((ext_vector_type(8)))  float    v8f;
typedef __attribute__((ext_vector_type(4)))  int      v4i;

typedef __attribute__((address_space(1))) v4i gv4i;   // global int4
typedef __attribute__((address_space(3))) v4i lv4i;   // LDS int4

#define BATCH 8
#define CIN   128
#define COUT  128
#define RES   256
#define WDIM  512
#define INV_SQRT_WDIM 0.04419417382415922f   // 1/sqrt(512)
#define LRELU_SLOPE 0.2f
#define LRELU_GAIN  1.4142135623730951f

// Activation strip staged in LDS: 3 rows x 66 px x 128 cin (f16),
// per-pixel stride padded to 136 halves (272 B = 17*16 B) so the 16 lanes of a
// half-wave read disjoint bank quads (4*lane mod 64 all distinct) -> conflict-free.
#define QPX  66
#define PSTR 136

#if __has_builtin(__builtin_amdgcn_global_load_async_to_lds_b128)
#define HAVE_ASYNC 1
#else
#define HAVE_ASYNC 0
#endif

#define SHUF16(lo, hi) __builtin_shufflevector(lo, hi, 0,1,2,3,4,5,6,7,8,9,10,11,12,13,14,15)

// ---------------------------------------------------------------------------
// Kernel 1: styles[b][i] = (w[b] . affine_w[i]) / sqrt(512) + affine_b[i]
// ---------------------------------------------------------------------------
__global__ void styles_kernel(const float* __restrict__ w,
                              const float* __restrict__ affine_w,
                              const float* __restrict__ affine_b,
                              float* __restrict__ styles) {
    int b = blockIdx.x;
    int i = threadIdx.x;
    float s = 0.f;
    const float* wr = w + (size_t)b * WDIM;
    const float* ar = affine_w + (size_t)i * WDIM;
    #pragma unroll 8
    for (int k = 0; k < WDIM; ++k) s += wr[k] * ar[k];
    styles[b * CIN + i] = s * INV_SQRT_WDIM + affine_b[i];
}

// ---------------------------------------------------------------------------
// Kernel 2: dcoef[b][o] = rsqrt( sum_i (sum_9 weight^2) * styles^2 + 1e-8 )
// ---------------------------------------------------------------------------
__global__ void dcoef_kernel(const float* __restrict__ weight,
                             const float* __restrict__ styles,
                             float* __restrict__ dcoef) {
    int t = blockIdx.x * blockDim.x + threadIdx.x;   // 0..1023
    int b = t / COUT;
    int o = t % COUT;
    float sum = 0.f;
    for (int i = 0; i < CIN; ++i) {
        const float* wp = weight + ((size_t)o * CIN + i) * 9;
        float w2 = 0.f;
        #pragma unroll
        for (int k = 0; k < 9; ++k) { float v = wp[k]; w2 += v * v; }
        float s = styles[b * CIN + i];
        sum += w2 * s * s;
    }
    dcoef[b * COUT + o] = rsqrtf(sum + 1e-8f);
}

// ---------------------------------------------------------------------------
// Kernel 3: weight transform OIHW f32 -> [kpos][o][i] f16 (cin contiguous)
// ---------------------------------------------------------------------------
__global__ void wt_kernel(const float* __restrict__ weight,
                          _Float16* __restrict__ wT) {
    int j = blockIdx.x * blockDim.x + threadIdx.x;   // 0..147455
    int i    = j % CIN;
    int o    = (j / CIN) % COUT;
    int kpos = j / (CIN * COUT);
    wT[j] = (_Float16)weight[((size_t)o * CIN + i) * 9 + kpos];
}

// ---------------------------------------------------------------------------
// Kernel 4: modulate + NCHW f32 -> NHWC f16 via LDS transpose (both sides coalesced)
// ---------------------------------------------------------------------------
__global__ void modulate_kernel(const float* __restrict__ x,
                                const float* __restrict__ styles,
                                _Float16* __restrict__ xm) {
    __shared__ float tile[CIN][33];
    int bid  = blockIdx.x;
    int xblk = bid & 7;
    int y    = (bid >> 3) & (RES - 1);
    int b    = bid >> 11;
    int t    = threadIdx.x;

    int xi = t & 31;
    int c0 = t >> 5;
    #pragma unroll
    for (int rep = 0; rep < 16; ++rep) {
        int c = rep * 8 + c0;
        float v = x[(((size_t)b * CIN + c) * RES + y) * RES + xblk * 32 + xi];
        tile[c][xi] = v * styles[b * CIN + c];
    }
    __syncthreads();

    int c = t & 127;
    int x0 = t >> 7;
    #pragma unroll
    for (int rep = 0; rep < 16; ++rep) {
        int xw = rep * 2 + x0;
        xm[(((size_t)b * RES + y) * RES + xblk * 32 + xw) * CIN + c] =
            (_Float16)tile[c][xw];
    }
}

// ---------------------------------------------------------------------------
// Kernel 5: implicit-GEMM 3x3 conv via v_wmma_f32_16x16x32_f16.
// WG = 8 waves = 4 M-waves x 2 N-waves; WG tile = 128 Cout x 64 px (fixed b,y).
// Activation strip (3 rows x 66 px x 128 cin, f16) staged once in LDS via
// async load-to-LDS (ASYNCcnt) when available; 9 taps x 4 Cin-chunks run
// entirely out of LDS. Each wave: 2x2 register tile -> 144 WMMAs.
// ---------------------------------------------------------------------------
__global__ void conv_wmma_kernel(const _Float16* __restrict__ xm,
                                 const _Float16* __restrict__ wT,
                                 const float* __restrict__ dcoef,
                                 const float* __restrict__ bias,
                                 const float* __restrict__ noise_const,
                                 const float* __restrict__ noise_strength,
                                 float* __restrict__ out) {
    __shared__ alignas(16) _Float16 strip[3 * QPX * PSTR];

    int bid   = blockIdx.x;
    int xtile = bid & 3;                // 64-pixel x tile
    int y     = (bid >> 2) & (RES - 1);
    int b     = bid >> 10;

    int tid  = threadIdx.x;
    int lane = tid & 31;
    int wave = tid >> 5;
    int half = lane >> 4;
    int ln   = lane & 15;
    int mw   = wave & 3;                // Cout block: rows [mw*32, mw*32+32)
    int nw   = wave >> 2;               // pixel block: [nw*32, nw*32+32) in tile

    // ---- Stage activation strip into LDS (zeros for padding halo) ----
    // strip is the only LDS object in this kernel -> LDS offset 0, so a raw
    // byte offset is a valid addrspace(3) address for the async-DMA builtin.
    for (int i = tid; i < 3 * QPX * 16; i += 256) {
        int r   = i / (QPX * 16);
        int rem = i % (QPX * 16);
        int q   = rem / 16;             // strip pixel 0..65 -> gx = xtile*64-1+q
        int c16 = rem % 16;             // 8-half (16 B) chunk of the 128 cin
        int yy  = y - 1 + r;
        int gx  = xtile * 64 - 1 + q;
        int loff = (r * QPX + q) * PSTR + c16 * 8;          // halves
        if (yy >= 0 && yy < RES && gx >= 0 && gx < RES) {
            size_t goff = (((size_t)b * RES + yy) * RES + gx) * CIN + c16 * 8;
#if HAVE_ASYNC
            __builtin_amdgcn_global_load_async_to_lds_b128(
                (gv4i*)(uintptr_t)(xm + goff),
                (lv4i*)(uintptr_t)(loff * 2),
                0, 0);
#else
            *(v8h*)&strip[loff] = *(const v8h*)(xm + goff);
#endif
        } else {
            v8h z = {};
            *(v8h*)&strip[loff] = z;
        }
    }
#if HAVE_ASYNC
#if __has_builtin(__builtin_amdgcn_s_wait_asynccnt)
    __builtin_amdgcn_s_wait_asynccnt(0);
#else
    asm volatile("s_wait_asynccnt 0" ::: "memory");
#endif
#endif
    __syncthreads();

    // ---- Main WMMA loop: 2x2 register tile per wave ----
    v8f a00 = {}, a01 = {}, a10 = {}, a11 = {};
    int mrow = mw * 32 + ln;

    for (int kh = 0; kh < 3; ++kh) {
        #pragma unroll
        for (int kw = 0; kw < 3; ++kw) {
            int kpos = kh * 3 + kw;
            #pragma unroll
            for (int kc = 0; kc < 4; ++kc) {
                // A fragments (weights): M rows mrow and mrow+16
                const _Float16* wp0 =
                    wT + ((size_t)(kpos * COUT + mrow) * CIN + kc * 32 + half * 8);
                v8h l0 = *(const v8h*)wp0;
                v8h h0 = *(const v8h*)(wp0 + 16);
                v16h af0 = SHUF16(l0, h0);
                const _Float16* wp1 = wp0 + 16 * CIN;
                v8h l1 = *(const v8h*)wp1;
                v8h h1 = *(const v8h*)(wp1 + 16);
                v16h af1 = SHUF16(l1, h1);

                // B fragments from LDS; shared by both M sub-tiles
                {
                    int q   = nw * 32 + 0 * 16 + ln + kw;
                    int off = (kh * QPX + q) * PSTR + kc * 32 + half * 8;
                    v8h bl = *(const v8h*)&strip[off];
                    v8h bh = *(const v8h*)&strip[off + 16];
                    v16h bf = SHUF16(bl, bh);
                    a00 = __builtin_amdgcn_wmma_f32_16x16x32_f16(
                        false, af0, false, bf, (short)0, a00, false, false);
                    a10 = __builtin_amdgcn_wmma_f32_16x16x32_f16(
                        false, af1, false, bf, (short)0, a10, false, false);
                }
                {
                    int q   = nw * 32 + 1 * 16 + ln + kw;
                    int off = (kh * QPX + q) * PSTR + kc * 32 + half * 8;
                    v8h bl = *(const v8h*)&strip[off];
                    v8h bh = *(const v8h*)&strip[off + 16];
                    v16h bf = SHUF16(bl, bh);
                    a01 = __builtin_amdgcn_wmma_f32_16x16x32_f16(
                        false, af0, false, bf, (short)0, a01, false, false);
                    a11 = __builtin_amdgcn_wmma_f32_16x16x32_f16(
                        false, af1, false, bf, (short)0, a11, false, false);
                }
            }
        }
    }

    // ---- Fused epilogue: demod, noise, bias, lrelu, gain, clamp -> NCHW f32 ----
    float nsv = noise_strength[0];
    #pragma unroll
    for (int ms = 0; ms < 2; ++ms) {
        #pragma unroll
        for (int ns2 = 0; ns2 < 2; ++ns2) {
            v8f acc = (ms == 0) ? (ns2 == 0 ? a00 : a01)
                                : (ns2 == 0 ? a10 : a11);
            int px = xtile * 64 + nw * 32 + ns2 * 16 + ln;
            float nz = noise_const[y * RES + px] * nsv;
            #pragma unroll
            for (int v = 0; v < 8; ++v) {
                int cout = mw * 32 + ms * 16 + 8 * half + v;  // C/D: M = v + 8*half
                float val = acc[v] * dcoef[b * COUT + cout] + nz + bias[cout];
                val = (val >= 0.f) ? val : LRELU_SLOPE * val;
                val *= LRELU_GAIN;
                val = fminf(fmaxf(val, -256.f), 256.f);
                out[(((size_t)b * COUT + cout) * RES + y) * RES + px] = val;
            }
        }
    }
}

// ---------------------------------------------------------------------------
extern "C" void kernel_launch(void* const* d_in, const int* in_sizes, int n_in,
                              void* d_out, int out_size, void* d_ws, size_t ws_size,
                              hipStream_t stream) {
    const float* x              = (const float*)d_in[0];
    const float* w              = (const float*)d_in[1];
    const float* affine_w       = (const float*)d_in[2];
    const float* affine_b       = (const float*)d_in[3];
    const float* weight         = (const float*)d_in[4];
    const float* noise_const    = (const float*)d_in[5];
    const float* noise_strength = (const float*)d_in[6];
    const float* bias           = (const float*)d_in[7];
    // d_in[8] = noise_mode — reference path is 'const'; ignored
    float* out = (float*)d_out;

    // Workspace layout:
    //   [0,      4096)   styles  f32 [8][128]
    //   [4096,   8192)   dcoef   f32 [8][128]
    //   [8192, 303104)   wT      f16 [9][128][128]
    //   [303104, +134MB) xm      f16 [8][256][256][128]  (NHWC)
    char* ws = (char*)d_ws;
    float*    styles = (float*)ws;
    float*    dcoef  = (float*)(ws + 4096);
    _Float16* wT     = (_Float16*)(ws + 8192);
    _Float16* xm     = (_Float16*)(ws + 303104);

    styles_kernel  <<<BATCH, CIN, 0, stream>>>(w, affine_w, affine_b, styles);
    dcoef_kernel   <<<4, 256, 0, stream>>>(weight, styles, dcoef);
    wt_kernel      <<<(9 * CIN * COUT) / 256, 256, 0, stream>>>(weight, wT);
    modulate_kernel<<<BATCH * RES * 8, 256, 0, stream>>>(x, styles, xm);
    conv_wmma_kernel<<<BATCH * RES * 4, 256, 0, stream>>>(
        xm, wT, dcoef, bias, noise_const, noise_strength, out);
}
